// GridVINR_61916248539357
// MI455X (gfx1250) — matrix operand order repaired and checked by reading, not compiled, not run on hardware
//
#include <hip/hip_runtime.h>

typedef __attribute__((ext_vector_type(16))) _Float16 v16h;
typedef __attribute__((ext_vector_type(8)))  float    v8f;
typedef __attribute__((ext_vector_type(4)))  unsigned u32x4;   // plain vector: works in as(3)

#define LDS __attribute__((address_space(3)))
#define Q_LEVELS 63.0f

// ---------------- fast branchless math (native trans ops) ----------------

static __device__ inline float fast_rcp(float x) {
#if __has_builtin(__builtin_amdgcn_rcpf)
  return __builtin_amdgcn_rcpf(x);
#else
  return 1.0f / x;
#endif
}

static __device__ inline float fast_exp2(float x) {
#if __has_builtin(__builtin_amdgcn_exp2f)
  return __builtin_amdgcn_exp2f(x);
#else
  return exp2f(x);
#endif
}

static __device__ inline float fast_tanh(float x) {
#if __has_builtin(__builtin_amdgcn_tanhf)
  return __builtin_amdgcn_tanhf(x);            // v_tanh_f32 (new on gfx1250)
#else
  float e = fast_exp2(fabsf(x) * -2.8853900817779268f);   // e^(-2|x|)
  float r = 1.0f - 2.0f * e * fast_rcp(1.0f + e);
  return copysignf(r, x);
#endif
}

// exact GELU via branchless Abramowitz-Stegun 7.1.26 erf (|err|<1.5e-7)
static __device__ inline float gelu_exact(float x) {
  float z  = x * 0.70710678118654752f;
  float az = fabsf(z);
  float t  = fast_rcp(fmaf(0.3275911f, az, 1.0f));
  float p  = fmaf(t, 1.061405429f, -1.453152027f);
  p = fmaf(t, p, 1.421413741f);
  p = fmaf(t, p, -0.284496736f);
  p = fmaf(t, p, 0.254829592f);
  p = p * t;
  float e    = fast_exp2(-az * az * 1.4426950408889634f); // e^(-z^2)
  float erfa = 1.0f - p * e;
  float erf  = copysignf(erfa, z);
  return 0.5f * x * (1.0f + erf);
}

// ---------------- WMMA helpers ----------------

static __device__ inline v8f wmma_f16(v16h a, v16h b, v8f c) {
  return __builtin_amdgcn_wmma_f32_16x16x32_f16(
      false, a, false, b, (short)0, c, false, false);
}

union HV16 { v16h v; unsigned u[8]; u32x4 q[2]; _Float16 h[16]; };

// Fragment load from LDS (addrspace(3)) -> two ds_load_b128, 32-bit addressing.
static __device__ inline v16h load_frag(const LDS _Float16* base, int slot) {
  const LDS u32x4* p = (const LDS u32x4*)(base + slot * 16);
  HV16 r;
  r.q[0] = p[0];
  r.q[1] = p[1];
  return r.v;
}

static __device__ inline void grid_accum(float coord, const float4* tab, int res,
                                         float& a0, float& a1, float& a2, float& a3) {
  float c   = (coord + 1.0f) * 0.5f * (float)(res - 1);
  int left  = min((int)floorf(c), res - 2);
  int right = max((int)ceilf(c), 1);
  float w   = c - (float)left;
  float4 L = tab[left];
  float4 R = tab[right];
  a0 += (1.0f - w) * L.x + w * R.x;
  a1 += (1.0f - w) * L.y + w * R.y;
  a2 += (1.0f - w) * L.z + w * R.z;
  a3 += (1.0f - w) * L.w + w * R.w;
}

// ---------------- prep: grid argmax tables ----------------

__global__ void prep_grids(const float* ind0, const float* cb0,
                           const float* ind1, const float* cb1,
                           const float* ind2, const float* cb2,
                           float* gtab) {
  int t = blockIdx.x * blockDim.x + threadIdx.x;
  if (t >= 2688) return;
  const float* ind; const float* cb; int row;
  if (t < 128)      { ind = ind0; cb = cb0; row = t; }
  else if (t < 640) { ind = ind1; cb = cb1; row = t - 128; }
  else              { ind = ind2; cb = cb2; row = t - 640; }
  const float* r = ind + row * 64;
  int arg = 0; float best = r[0];
#pragma unroll 8
  for (int j = 1; j < 64; ++j) {
    float v = r[j];
    if (v > best) { best = v; arg = j; }
  }
  ((float4*)gtab)[t] = ((const float4*)cb)[arg];
}

// ---------------- prep: quantize + pre-permute into WMMA frag images ----------------

struct PW { const float* src[10]; };

__global__ void prep_weights(PW pw, float* bias, _Float16* Wh) {
  const int sizes[10] = {320, 4096, 4096, 4096, 64, 64, 64, 64, 64, 1};
  int b   = blockIdx.x;
  int tid = threadIdx.x;
  const float* src = pw.src[b];
  int sz = sizes[b];

  __shared__ float smn[256], smx[256];
  float mn = 3.4e38f, mx = -3.4e38f;
  for (int i = tid; i < sz; i += 256) {
    float v = src[i];
    mn = fminf(mn, v); mx = fmaxf(mx, v);
  }
  smn[tid] = mn; smx[tid] = mx;
  __syncthreads();
  for (int s = 128; s > 0; s >>= 1) {
    if (tid < s) {
      smn[tid] = fminf(smn[tid], smn[tid + s]);
      smx[tid] = fmaxf(smx[tid], smx[tid + s]);
    }
    __syncthreads();
  }
  mn = smn[0]; mx = smx[0];
  float scale = fmaxf(mx - mn, 1e-8f) / Q_LEVELS;
  float inv   = 1.0f / scale;
#define QQ(v) (rintf(((v) - mn) * inv) * scale + mn)

  if (b == 0) {
    // w0 (5x64) -> 4 B-frags (kf=0 only), zero-padded K 5..31
    for (int j = tid; j < 2048; j += 256) {
      int nt = j >> 9, lane = (j >> 4) & 31, pos = j & 15;
      int n = nt * 16 + (lane & 15);
      int k = ((lane & 16) ? 16 : 0) + pos;
      float q = (k < 5) ? QQ(src[k * 64 + n]) : 0.0f;
      Wh[j] = (_Float16)q;
    }
  } else if (b <= 3) {
    // w1..w3 (64x64) -> 8 B-frags each (nt,kf)
    _Float16* dst = Wh + 2048 + (b - 1) * 4096;
    for (int j = tid; j < 4096; j += 256) {
      int fb = j >> 9, lane = (j >> 4) & 31, pos = j & 15;
      int nt = fb >> 1, kf = fb & 1;
      int n = nt * 16 + (lane & 15);
      int k = kf * 32 + ((lane & 16) ? 8 : 0) + (pos & 7) + ((pos & 8) ? 16 : 0);
      dst[j] = (_Float16)QQ(src[k * 64 + n]);
    }
  } else if (b == 4) {
    // w4 (64) permuted to match activation fragment-image chunk order
    _Float16* dst = Wh + 14336;
    for (int j = tid; j < 64; j += 256) {
      int kf = j >> 5, half = (j >> 4) & 1, pos = j & 15;
      int v = pos >> 1, h = pos & 1;
      int k = kf * 32 + half * 8 + 2 * v + h + (v >= 4 ? 8 : 0);
      dst[j] = (_Float16)QQ(src[k]);
    }
  } else if (b <= 8) {
    float* dst = bias + (b - 5) * 64;
    for (int j = tid; j < 64; j += 256) dst[j] = QQ(src[j]);
  } else {
    if (tid == 0) bias[256] = QQ(src[0]);
  }
#undef QQ
}

// ---------------- main fused MLP kernel ----------------

#define WAVES 8

// 64->64 layer, all LDS traffic via addrspace(3) pointers.
static __device__ inline void layer64(LDS _Float16* scr, const LDS _Float16* W,
                                      const float* bb /*4 regs*/, int lane) {
  v16h a[2][2];
#pragma unroll
  for (int mt = 0; mt < 2; ++mt)
#pragma unroll
    for (int kf = 0; kf < 2; ++kf)
      a[mt][kf] = load_frag(scr, (mt * 2 + kf) * 32 + lane);
  asm volatile("" ::: "memory");

  int half = (lane >> 3) & 1;
  int hbit = lane & 1;
  int rl0  = (lane & 16) ? 8 : 0;

#pragma unroll
  for (int nt = 0; nt < 4; ++nt) {
    v16h bl = load_frag(W, (nt * 2 + 0) * 32 + lane);
    v16h bh = load_frag(W, (nt * 2 + 1) * 32 + lane);
    int kf = nt >> 1;
    int v  = ((lane & 7) >> 1) + (nt & 1) * 4;
#pragma unroll
    for (int mt = 0; mt < 2; ++mt) {
      v8f d = {};
      d = wmma_f16(a[mt][0], bl, d);
      d = wmma_f16(a[mt][1], bh, d);
      int sbase = ((mt * 2 + kf) * 32 + rl0 + 16 * half) * 16 + v * 2 + hbit;
#pragma unroll
      for (int i = 0; i < 8; ++i) {
        float xv = gelu_exact(d[i] + bb[nt]);
        scr[sbase + i * 16] = (_Float16)xv;
      }
    }
  }
  asm volatile("" ::: "memory");
}

__global__ __launch_bounds__(256) void mlp_kernel(const float* __restrict__ x,
                                                  const float* __restrict__ gtab,
                                                  const float* __restrict__ bias,
                                                  const _Float16* __restrict__ Wg,
                                                  float* __restrict__ out,
                                                  int npts) {
  __shared__ __align__(16) _Float16 sW[14400];
  __shared__ float sB[257];
  __shared__ __align__(16) _Float16 sScr[WAVES][32 * 64];

  {
    const unsigned* src = (const unsigned*)Wg;
    unsigned* dst = (unsigned*)sW;
    for (int i = threadIdx.x; i < 7200; i += 256) dst[i] = src[i];
    for (int i = threadIdx.x; i < 257; i += 256) sB[i] = bias[i];
  }
  __syncthreads();

  int lane = threadIdx.x & 31;
  int wave = threadIdx.x >> 5;

  // explicit addrspace(3) views: 32-bit DS addressing, no generic-pointer casts
  const LDS _Float16* sWp = (const LDS _Float16*)sW;
  const LDS float*    sBp = (const LDS float*)sB;
  LDS _Float16*       scr = (LDS _Float16*)&sScr[wave][0];

  const LDS _Float16* W0 = sWp;           // 4 frags
  const LDS _Float16* W1 = sWp + 2048;    // 8 frags
  const LDS _Float16* W2 = sWp + 6144;
  const LDS _Float16* W3 = sWp + 10240;
  const LDS _Float16* W4 = sWp + 14336;   // 4 chunks of 16 (permuted)

  // loop-invariant per-lane constants: w4 fragments + biases in registers
  HV16 w4c[4];
#pragma unroll
  for (int c = 0; c < 4; ++c) w4c[c].v = load_frag(W4, c);
  float bb[4][4];
#pragma unroll
  for (int l = 0; l < 4; ++l)
#pragma unroll
    for (int nt = 0; nt < 4; ++nt)
      bb[l][nt] = sBp[l * 64 + nt * 16 + (lane & 15)];
  float b4v = sBp[256];

  const float4* tab0 = (const float4*)gtab;
  const float4* tab1 = (const float4*)gtab + 128;
  const float4* tab2 = (const float4*)gtab + 640;
  const float2* x2   = (const float2*)x;

  int ntiles  = npts >> 5;            // 32 points per wave-tile
  int tstride = gridDim.x * WAVES;

  for (int tile = blockIdx.x * WAVES + wave; tile < ntiles; tile += tstride) {
    int p = tile * 32 + lane;
    float2 xv = x2[p];                                   // global_load_b64
    float feat  = xv.x;
    float coord = xv.y;
    __builtin_prefetch(&x2[p + tstride * 32], 0, 0);     // global_prefetch_b8

    float g0 = 0.f, g1 = 0.f, g2 = 0.f, g3 = 0.f;
    grid_accum(coord, tab0, 128,  g0, g1, g2, g3);
    grid_accum(coord, tab1, 512,  g0, g1, g2, g3);
    grid_accum(coord, tab2, 2048, g0, g1, g2, g3);

    // ---- build h0 frag image: zero kf=0 slots, then 5 live halves ----
    {
      u32x4 z = {0u, 0u, 0u, 0u};
      LDS u32x4* s4 = (LDS u32x4*)scr;
      s4[(0 * 32 + lane) * 2 + 0] = z;   // fid 0 (mt=0,kf=0)
      s4[(0 * 32 + lane) * 2 + 1] = z;
      s4[(2 * 32 + lane) * 2 + 0] = z;   // fid 2 (mt=1,kf=0)
      s4[(2 * 32 + lane) * 2 + 1] = z;
      int fid = (lane >> 4) * 2;
      LDS _Float16* slot = scr + (fid * 32 + (lane & 15)) * 16;
      slot[0] = (_Float16)g0;
      slot[1] = (_Float16)g1;
      slot[2] = (_Float16)g2;
      slot[3] = (_Float16)g3;
      slot[4] = (_Float16)feat;
    }
    asm volatile("" ::: "memory");

    // ---- layer 0: single k-step ----
    {
      v16h a0 = load_frag(scr, 0 * 32 + lane);
      v16h a1 = load_frag(scr, 2 * 32 + lane);
      asm volatile("" ::: "memory");
      int half = (lane >> 3) & 1;
      int hbit = lane & 1;
      int rl0  = (lane & 16) ? 8 : 0;
#pragma unroll
      for (int nt = 0; nt < 4; ++nt) {
        v16h bfr = load_frag(W0, nt * 32 + lane);
        int kf = nt >> 1;
        int v  = ((lane & 7) >> 1) + (nt & 1) * 4;
#pragma unroll
        for (int mt = 0; mt < 2; ++mt) {
          v8f d = {};
          d = wmma_f16(mt == 0 ? a0 : a1, bfr, d);
          int sbase = ((mt * 2 + kf) * 32 + rl0 + 16 * half) * 16 + v * 2 + hbit;
#pragma unroll
          for (int i = 0; i < 8; ++i) {
            float xg = gelu_exact(d[i] + bb[0][nt]);
            scr[sbase + i * 16] = (_Float16)xg;
          }
        }
      }
      asm volatile("" ::: "memory");
    }

    // ---- layers 1..3 ----
    layer64(scr, W1, bb[1], lane);
    layer64(scr, W2, bb[2], lane);
    layer64(scr, W3, bb[3], lane);

    // ---- layer 4: lane reads its own row as 4 contiguous frag chunks ----
    {
      float sum = 0.0f;
      int mt = lane >> 4;
#pragma unroll
      for (int kf = 0; kf < 2; ++kf) {
#pragma unroll
        for (int hf = 0; hf < 2; ++hf) {
          HV16 r;
          r.v = load_frag(scr, (mt * 2 + kf) * 32 + (lane & 15) + 16 * hf);
          const HV16& w = w4c[kf * 2 + hf];
#pragma unroll
          for (int tix = 0; tix < 16; ++tix)
            sum += (float)r.h[tix] * (float)w.h[tix];
        }
      }
      out[p] = fast_tanh(sum + b4v);
    }
  }
}

// ---------------- launch ----------------

extern "C" void kernel_launch(void* const* d_in, const int* in_sizes, int n_in,
                              void* d_out, int out_size, void* d_ws, size_t ws_size,
                              hipStream_t stream) {
  const float* cb0  = (const float*)d_in[1];
  const float* ind0 = (const float*)d_in[2];
  const float* cb1  = (const float*)d_in[3];
  const float* ind1 = (const float*)d_in[4];
  const float* cb2  = (const float*)d_in[5];
  const float* ind2 = (const float*)d_in[6];

  char* ws = (char*)d_ws;
  float*    gtab = (float*)ws;                 // 43008 B
  float*    bias = (float*)(ws + 43008);       // 257 f32
  _Float16* Wh   = (_Float16*)(ws + 44048);    // 14400 f16

  int npts = in_sizes[0] / 2;

  prep_grids<<<(2688 + 255) / 256, 256, 0, stream>>>(ind0, cb0, ind1, cb1, ind2, cb2, gtab);

  PW pw;
  pw.src[0] = (const float*)d_in[7];
  pw.src[1] = (const float*)d_in[9];
  pw.src[2] = (const float*)d_in[11];
  pw.src[3] = (const float*)d_in[13];
  pw.src[4] = (const float*)d_in[15];
  pw.src[5] = (const float*)d_in[8];
  pw.src[6] = (const float*)d_in[10];
  pw.src[7] = (const float*)d_in[12];
  pw.src[8] = (const float*)d_in[14];
  pw.src[9] = (const float*)d_in[16];
  prep_weights<<<10, 256, 0, stream>>>(pw, bias, Wh);

  mlp_kernel<<<1024, 256, 0, stream>>>((const float*)d_in[0], gtab, bias, Wh,
                                       (float*)d_out, npts);
}
